// AdditionNetwork_1_51101520888076
// MI455X (gfx1250) — matrix operand order repaired and checked by reading, not compiled, not run on hardware
//
#include <hip/hip_runtime.h>
#include <stdint.h>

typedef __attribute__((ext_vector_type(16))) _Float16 v16h;
typedef __attribute__((ext_vector_type(8)))  float    v8f;

#define DH 15
#define DW 20
#define NREAL 300   // real token count 15*20
#define NPAD  320   // K-dim padding (multiple of 32)
#define NROW  304   // padded output rows (19 tiles of 16)
#define BATCH 8

// ---------------------------------------------------------------------------
// Prep Q: depthwise conv (stride K, VALID) of rgb -> f16 q[b][n][c], n<320,
// rows >= 300 zero-padded so WMMA K/col tiles can read them safely.
// Thread index has n fastest: a wave sweeps adjacent K-wide patches, so the
// fully-unrolled row reads coalesce into contiguous b128 loads (this pass
// reads all of rgb once -> it is bandwidth-bound).
// ---------------------------------------------------------------------------
template <int K>
__global__ void prep_q_kernel(const float* __restrict__ rgb,
                              const float* __restrict__ wgt,
                              _Float16* __restrict__ q,
                              int C, int H, int W) {
  int idx = blockIdx.x * blockDim.x + threadIdx.x;
  int total = BATCH * NPAD * C;
  if (idx >= total) return;
  int n = idx % NPAD;
  int c = (idx / NPAD) % C;
  int b = idx / (NPAD * C);
  float acc = 0.0f;
  if (n < NREAL) {
    int oh = n / DW, ow = n % DW;
    const float* rp = rgb + ((size_t)(b * C + c) * H + (size_t)oh * K) * W + (size_t)ow * K;
    const float* wp = wgt + (size_t)c * K * K;
#pragma unroll
    for (int i = 0; i < K; ++i)
#pragma unroll
      for (int j = 0; j < K; ++j)
        acc += rp[(size_t)i * W + j] * wp[i * K + j];
  }
  q[((size_t)b * NPAD + n) * C + c] = (_Float16)acc;
}

// ---------------------------------------------------------------------------
// Prep V: nearest-downsample depth -> f16 vT[b][c][n] (transposed so B
// fragments read contiguous K pairs), n<320 with zero pad. Reads are sparse
// gathers (only 300 of H*W pixels per plane) -> negligible traffic.
// ---------------------------------------------------------------------------
__global__ void prep_v_kernel(const float* __restrict__ depth,
                              _Float16* __restrict__ vt,
                              int C, int H, int W) {
  int idx = blockIdx.x * blockDim.x + threadIdx.x;
  int total = BATCH * C * NPAD;
  if (idx >= total) return;
  int n = idx % NPAD;
  int c = (idx / NPAD) % C;
  int b = idx / (NPAD * C);
  float val = 0.0f;
  if (n < NREAL) {
    int oh = n / DW, ow = n % DW;
    int ih = oh * (H / DH);   // exact: H is a multiple of 15
    int iw = ow * (W / DW);   // exact: W is a multiple of 20
    val = depth[((size_t)(b * C + c) * H + ih) * W + iw];
  }
  vt[((size_t)b * C + c) * NPAD + n] = (_Float16)val;
}

// ---------------------------------------------------------------------------
// Fused attention: one block = (batch b, row tile of 16 query rows).
//   phase 1: aff[16 x 320] = q_tile . q^T   (WMMA f16 -> f32, K = C)
//            + key mask (-1e9 for masked real cols, -3e9 for pad cols)
//   phase 2: row softmax (f32, lane-shuffle reductions), -> f16 in LDS
//   phase 3: dn[16 x C] = aff_h . v         (WMMA f16 -> f32, K = 320)
// 8 waves / block; all WMMA in wave-uniform control flow (EXEC all ones).
// ---------------------------------------------------------------------------
__launch_bounds__(256, 1)
__global__ void attn_kernel(const _Float16* __restrict__ q,
                            const _Float16* __restrict__ vt,
                            const int* __restrict__ mask,
                            float* __restrict__ dn, int C) {
  __shared__ float    affF[16 * NPAD];   // 20 KB
  __shared__ _Float16 affH[16 * NPAD];   // 10 KB

  const int tid  = threadIdx.x;
  const int lane = tid & 31;
  const int wave = tid >> 5;
  const int half = lane >> 4;
  const int l15  = lane & 15;

  const int rt       = blockIdx.x % 19;
  const int b        = blockIdx.x / 19;
  const int row_base = rt * 16;

  const _Float16* qb = q  + (size_t)b * NPAD * C;
  const _Float16* vb = vt + (size_t)b * C * NPAD;

  // ---- phase 1: Q . Q^T column tiles --------------------------------------
  const _Float16* arow = qb + (size_t)(row_base + l15) * C;  // A: M = lane&15
  for (int ct = wave; ct < NPAD / 16; ct += 8) {
    const int col_base = ct * 16;
    const _Float16* brow = qb + (size_t)(col_base + l15) * C; // B: N = lane&15
    v8f acc = {};
    for (int kk = 0; kk < C; kk += 32) {
      union { v16h h; uint32_t u[8]; } A, Bf;
#pragma unroll
      for (int v = 0; v < 8; ++v) {
        const int ka = kk + 2 * (v & 3) + 8 * half + ((v & 4) ? 16 : 0);
        A.u[v]  = *(const uint32_t*)(arow + ka);
        const int kb = kk + 2 * (v & 3) + 8 * half + ((v & 4) ? 16 : 0);
        Bf.u[v] = *(const uint32_t*)(brow + kb);
      }
      acc = __builtin_amdgcn_wmma_f32_16x16x32_f16(
          false, A.h, false, Bf.h, (short)0, acc, false, false);
    }
    // key-column mask (branchless per lane; one column per lane)
    const int col = col_base + l15;
    const int cc  = (col < NREAL) ? col : 0;
    const int mh  = ((cc / DW) * 8) / 15;
    const int mw  = ((cc % DW) * 8) / 20;
    const int mv  = mask[(b * 8 + mh) * 8 + mw];
    const bool real  = (col < NREAL);
    const bool keep  = real && (mv != 0);
#pragma unroll
    for (int r = 0; r < 8; ++r) {
      float x = keep ? acc[r] : (real ? -1e9f : -3e9f);
      affF[(r + 8 * half) * NPAD + col] = x;
    }
  }
  __syncthreads();

  // ---- phase 2: row softmax (each wave owns 2 rows) -----------------------
  for (int rr = 0; rr < 2; ++rr) {
    const int row = wave * 2 + rr;
    float x[NPAD / 32];
    float m = -3.4e38f;
#pragma unroll
    for (int i = 0; i < NPAD / 32; ++i) {
      x[i] = affF[row * NPAD + lane + 32 * i];
      m = fmaxf(m, x[i]);
    }
#pragma unroll
    for (int off = 16; off > 0; off >>= 1) m = fmaxf(m, __shfl_xor(m, off, 32));
    float s = 0.0f;
#pragma unroll
    for (int i = 0; i < NPAD / 32; ++i) {
      x[i] = expf(x[i] - m);
      s += x[i];
    }
#pragma unroll
    for (int off = 16; off > 0; off >>= 1) s += __shfl_xor(s, off, 32);
    const float inv = 1.0f / s;
#pragma unroll
    for (int i = 0; i < NPAD / 32; ++i)
      affH[row * NPAD + lane + 32 * i] = (_Float16)(x[i] * inv);
  }
  __syncthreads();

  // ---- phase 3: aff_h . V  (K = 320, A from LDS) --------------------------
  for (int ct = wave; ct < C / 16; ct += 8) {
    const int c_base = ct * 16;
    const _Float16* bcol = vb + (size_t)(c_base + l15) * NPAD; // B: N = lane&15
    v8f acc = {};
    for (int kk = 0; kk < NPAD; kk += 32) {
      union { v16h h; uint32_t u[8]; } A, Bf;
#pragma unroll
      for (int v = 0; v < 8; ++v) {
        const int ka = kk + 2 * (v & 3) + 8 * half + ((v & 4) ? 16 : 0);
        A.u[v]  = *(const uint32_t*)&affH[l15 * NPAD + ka];
        const int kb = kk + 2 * (v & 3) + 8 * half + ((v & 4) ? 16 : 0);
        Bf.u[v] = *(const uint32_t*)(bcol + kb);
      }
      acc = __builtin_amdgcn_wmma_f32_16x16x32_f16(
          false, A.h, false, Bf.h, (short)0, acc, false, false);
    }
    // D layout: lane holds column c_base+l15, rows r + 8*half -> dn[b][c][n]
    float* drow = dn + ((size_t)b * C + c_base + l15) * NROW + row_base;
#pragma unroll
    for (int r = 0; r < 8; ++r) drow[r + 8 * half] = acc[r];
  }
}

// ---------------------------------------------------------------------------
// Upsample (nearest, exact integer ratio) + residual add, float4-vectorized.
// This is the HBM-bound bulk of the op (~0.46 GB of the ~0.7 GB total).
// W is a multiple of 4 at every level.
// ---------------------------------------------------------------------------
__global__ void upsample_add_kernel(const float4* __restrict__ rgb4,
                                    const float* __restrict__ dn,
                                    float4* __restrict__ out4,
                                    int C, int H, int W) {
  const size_t total4 = (size_t)BATCH * C * H * (W / 4);
  size_t idx = (size_t)blockIdx.x * blockDim.x + threadIdx.x;
  if (idx >= total4) return;
  const int wq = W / 4;
  const int w0 = (int)(idx % wq) * 4;
  const int h  = (int)((idx / wq) % H);
  const size_t bc = idx / ((size_t)wq * H);         // b*C + c
  const int wd = W / DW;
  const float* drow = dn + bc * NROW + (size_t)(h / (H / DH)) * DW;
  const float4 r = rgb4[idx];
  float4 o;
  o.x = r.x + drow[(w0 + 0) / wd];
  o.y = r.y + drow[(w0 + 1) / wd];
  o.z = r.z + drow[(w0 + 2) / wd];
  o.w = r.w + drow[(w0 + 3) / wd];
  out4[idx] = o;
}

// ---------------------------------------------------------------------------
extern "C" void kernel_launch(void* const* d_in, const int* in_sizes, int n_in,
                              void* d_out, int out_size, void* d_ws, size_t ws_size,
                              hipStream_t stream) {
  (void)in_sizes; (void)n_in; (void)out_size; (void)ws_size;
  static const int Cs[5] = {64, 64, 128, 256, 512};
  static const int Hs[5] = {240, 120, 60, 30, 15};
  static const int Ws[5] = {320, 160, 80, 40, 20};

  // scratch layout (reused level to level; stream-ordered):
  //   q  : f16 [B][320][512max]  = 2.50 MB
  //   vT : f16 [B][512max][320]  = 2.50 MB
  //   dn : f32 [B][512max][304]  = 4.75 MB
  char* ws = (char*)d_ws;
  _Float16* q  = (_Float16*)ws;
  _Float16* vt = (_Float16*)(ws + (size_t)BATCH * NPAD * 512 * 2);
  float*    dn = (float*)   (ws + (size_t)BATCH * NPAD * 512 * 2 * 2);

  const int* mask = (const int*)d_in[15];
  float* out = (float*)d_out;
  size_t out_off = 0;

  for (int l = 0; l < 5; ++l) {
    const float* rgb = (const float*)d_in[3 * l + 0];
    const float* dep = (const float*)d_in[3 * l + 1];
    const float* wgt = (const float*)d_in[3 * l + 2];
    const int C = Cs[l], H = Hs[l], W = Ws[l];

    const int tprep = BATCH * NPAD * C;
    const int gprep = (tprep + 255) / 256;
    switch (l) {
      case 0: prep_q_kernel<16><<<gprep, 256, 0, stream>>>(rgb, wgt, q, C, H, W); break;
      case 1: prep_q_kernel< 8><<<gprep, 256, 0, stream>>>(rgb, wgt, q, C, H, W); break;
      case 2: prep_q_kernel< 4><<<gprep, 256, 0, stream>>>(rgb, wgt, q, C, H, W); break;
      case 3: prep_q_kernel< 2><<<gprep, 256, 0, stream>>>(rgb, wgt, q, C, H, W); break;
      default: prep_q_kernel<1><<<gprep, 256, 0, stream>>>(rgb, wgt, q, C, H, W); break;
    }
    prep_v_kernel<<<gprep, 256, 0, stream>>>(dep, vt, C, H, W);

    attn_kernel<<<BATCH * 19, 256, 0, stream>>>(q, vt, mask, dn, C);

    const size_t tout4 = (size_t)BATCH * C * H * (W / 4);
    upsample_add_kernel<<<(unsigned)((tout4 + 255) / 256), 256, 0, stream>>>(
        (const float4*)rgb, dn, (float4*)(out + out_off), C, H, W);
    out_off += tout4 * 4;
  }
}
